// QuantLinear_34136400068881
// MI455X (gfx1250) — compile-verified, matrix operand order checked by reading, not run on hardware
//
#include <hip/hip_runtime.h>

// ---- CDNA5 WMMA vector types ----
typedef __attribute__((ext_vector_type(16))) __bf16 v16bf;
typedef __attribute__((ext_vector_type(8)))  float  v8f;
typedef __attribute__((ext_vector_type(2)))  int    v2i;
typedef __attribute__((ext_vector_type(2)))  float  v2f;

constexpr int IN_F    = 4096;
constexpr int OUT_F   = 4096;
constexpr int M_TOT   = 512;   // 4 * 128
constexpr int KB      = 32;    // K depth of v_wmma_f32_16x16x32_bf16
constexpr int NT      = 32;    // N tile per workgroup (weight rows read once)
constexpr int MW      = 32;    // M rows per wave (2 fragments)
constexpr int THREADS = 512;   // 16 waves * 32 lanes -> covers M = 512

constexpr int LDS_STRIDE = KB + 2;  // 34 bf16 = 17 dwords/row: gcd(17,64)=1 -> no bank conflicts

__global__ __launch_bounds__(THREADS, 2)
void quantlinear_wmma(const float* __restrict__ x,        // [512, 4096] f32
                      const int*   __restrict__ qw,       // [4096, 4096] i32 (0..15)
                      const float* __restrict__ scales,   // [4096]
                      const float* __restrict__ zeros,    // [4096]
                      const float* __restrict__ outlier,  // [4096, 4096]
                      const float* __restrict__ grad,     // [4096, 4096]
                      const float* __restrict__ bias,     // [4096]
                      float*       __restrict__ y)        // [512, 4096]
{
    // Double-buffered dequantized weight tile: NT x KB bf16
    __shared__ __bf16 Wl[2][NT * LDS_STRIDE];

    const int tid  = threadIdx.x;
    const int wave = tid >> 5;
    const int lane = tid & 31;
    const int l16  = lane & 15;
    const int hi   = lane >> 4;      // lane group 0 / 1

    const int ntile = blockIdx.x * NT;
    const int mbase = wave * MW;

    // Cooperative dequant mapping: 2 consecutive elements of the NT x KB tile per thread
    const int  dq_n = tid >> 4;            // 0..31 (weight row within tile)
    const int  dq_k = (tid & 15) * 2;      // 0,2,..,30
    const float sc  = scales[ntile + dq_n];
    const float zp  = zeros [ntile + dq_n];
    const long  wrow = (long)(ntile + dq_n) * IN_F;

    // 16-bit WMMA A/B fragment K-offsets per ISA layout:
    // lanes 0-15 hold K in {0..7} and {16..23}; lanes 16-31 hold {8..15} and {24..31}
    const int ka = hi * 8;
    const int kb = 16 + hi * 8;

    v8f acc[2][2] = {};   // [mf][nf] 16x16 f32 C tiles

    for (int k0 = 0; k0 < IN_F; k0 += KB) {
        const int buf = (k0 >> 5) & 1;

        // ---- fused dequant: W = (q - zero)*scale + outlier*grad  -> bf16 LDS tile ----
        // qweight/outlier/grad are streamed exactly once -> non-temporal (keep L2 for x)
        {
            const long base = wrow + k0 + dq_k;
            const v2i q = __builtin_nontemporal_load((const v2i*)(qw      + base));
            const v2f o = __builtin_nontemporal_load((const v2f*)(outlier + base));
            const v2f g = __builtin_nontemporal_load((const v2f*)(grad    + base));
            const float w0 = ((float)q.x - zp) * sc + o.x * g.x;
            const float w1 = ((float)q.y - zp) * sc + o.y * g.y;
            Wl[buf][dq_n * LDS_STRIDE + dq_k + 0] = (__bf16)w0;
            Wl[buf][dq_n * LDS_STRIDE + dq_k + 1] = (__bf16)w1;
        }
        __syncthreads();   // double buffer => one barrier per step is sufficient

        // ---- B fragments (32x16 bf16) from LDS: lane = column n, K striped per ISA map ----
        v16bf B[2];
        #pragma unroll
        for (int nf = 0; nf < 2; ++nf) {
            const __bf16* r = &Wl[buf][(nf * 16 + l16) * LDS_STRIDE];
            #pragma unroll
            for (int j = 0; j < 8; ++j) {
                B[nf][j]     = r[ka + j];
                B[nf][8 + j] = r[kb + j];
            }
        }

        // ---- A fragments (16x32 bf16) straight from global f32 x (L2-resident) ----
        #pragma unroll
        for (int mf = 0; mf < 2; ++mf) {
            const float* xr = x + (long)(mbase + mf * 16 + l16) * IN_F + k0;
            v16bf A;
            #pragma unroll
            for (int j = 0; j < 8; ++j) {
                A[j]     = (__bf16)xr[ka + j];
                A[8 + j] = (__bf16)xr[kb + j];
            }
            #pragma unroll
            for (int nf = 0; nf < 2; ++nf) {
                acc[mf][nf] = __builtin_amdgcn_wmma_f32_16x16x32_bf16(
                    /*neg_a=*/false, A, /*neg_b=*/false, B[nf],
                    /*c_mod=*/(short)0, acc[mf][nf],
                    /*reuse_a=*/false, /*reuse_b=*/false);
            }
        }
    }

    // ---- epilogue: C layout (VGPR r -> M=r | r+8 by lane group; lane -> N), add bias ----
    // y is written once -> non-temporal stores
    #pragma unroll
    for (int nf = 0; nf < 2; ++nf) {
        const int   n  = ntile + nf * 16 + l16;
        const float bn = bias[n];
        #pragma unroll
        for (int mf = 0; mf < 2; ++mf) {
            #pragma unroll
            for (int r = 0; r < 8; ++r) {
                const int m = mbase + mf * 16 + r + hi * 8;
                __builtin_nontemporal_store(acc[mf][nf][r] + bn, &y[(long)m * OUT_F + n]);
            }
        }
    }
}

extern "C" void kernel_launch(void* const* d_in, const int* in_sizes, int n_in,
                              void* d_out, int out_size, void* d_ws, size_t ws_size,
                              hipStream_t stream) {
    const float* x  = (const float*)d_in[0];   // input
    const int*   qw = (const int*)  d_in[1];   // qweight
    const float* sc = (const float*)d_in[2];   // scales
    const float* zp = (const float*)d_in[3];   // zeros
    const float* ol = (const float*)d_in[4];   // outlier
    const float* gu = (const float*)d_in[5];   // grad_updates
    const float* bs = (const float*)d_in[6];   // bias
    float* y = (float*)d_out;

    dim3 grid(OUT_F / NT);   // 128 workgroups, each owns full M=512 (weights read once)
    dim3 block(THREADS);     // 16 waves
    hipLaunchKernelGGL(quantlinear_wmma, grid, block, 0, stream,
                       x, qw, sc, zp, ol, gu, bs, y);
}
